// Poolformer_16432544874781
// MI455X (gfx1250) — compile-verified, matrix operand order checked by reading, not run on hardware
//
#include <hip/hip_runtime.h>
#include <hip/hip_bf16.h>

// ---------------------------------------------------------------------------
// Poolformer / long-short transformer forward for MI455X (gfx1250).
// Heavy GEMMs + attention cores: v_wmma_f32_16x16x32_bf16 (wave32).
// bf16 tile staging uses gfx1250 async global->LDS copies (ASYNCcnt).
// ---------------------------------------------------------------------------

#define DIMM    768
#define HEADS   12
#define DH      64
#define INNER   768
#define GT_H    8
#define GT_IN   512
#define WIN     128
#define DEPTH_L 4
#define SEQ_N   4096
#define BKT     32     // SEQ/WIN
#define FFD     3072

typedef __attribute__((ext_vector_type(16))) __bf16 v16bf;
typedef __attribute__((ext_vector_type(8)))  float  v8f;

// float -> bf16 (round-to-nearest-even) as raw bits
__device__ __forceinline__ unsigned short f2bfbits(float f) {
    union { float f; unsigned u; } c; c.f = f;
    return (unsigned short)((c.u + 0x7FFFu + ((c.u >> 16) & 1u)) >> 16);
}
__device__ __forceinline__ __bf16 f2bf(float f) {
    unsigned short s = f2bfbits(f);
    __bf16 o; __builtin_memcpy(&o, &s, 2); return o;
}

// WMMA bf16 wrapper: D = A(16x32) * B(32x16) + C (f32 accum)
__device__ __forceinline__ v8f bwmma(v16bf a, v16bf b, v8f c) {
    return __builtin_amdgcn_wmma_f32_16x16x32_bf16(
        false, a, false, b, (short)0, c, false, false);
}

// Async copy of 16 bytes global -> LDS (gfx1250, tracked by ASYNCcnt).
// ldsp: pointer into __shared__ (low 32 bits of the generic address are the
// workgroup-relative LDS offset); gp: global source.
__device__ __forceinline__ void async_g2l_b128(void* ldsp, const void* gp) {
    unsigned loff = (unsigned)(unsigned long long)ldsp;
    asm volatile("global_load_async_to_lds_b128 %0, %1, off"
                 :: "v"(loff), "v"(gp) : "memory");
}
__device__ __forceinline__ void async_wait0() {
    asm volatile("s_wait_asynccnt 0x0" ::: "memory");
}

// Load a 16x32 bf16 fragment from an LDS tile stored row-major with `pitch`
// elements per row (pitch multiple of 8; tile base 32B aligned).
// A-matrix layout (ISA 7.12.2): lanes 0-15 hold row=lane, K {0..7,16..23};
// lanes 16-31 hold row=lane-16, K {8..15,24..31}. B fragments use the
// mirrored layout (row := output column) from an N-major LDS tile.
__device__ __forceinline__ v16bf ldsFrag(const __bf16* tile, int pitch, int lane) {
    int l = lane & 15, hf = lane >> 4;
    const __bf16* p = tile + l * pitch + hf * 8;
    union { v16bf v; uint4 u[2]; } f;
    f.u[0] = *(const uint4*)(p);
    f.u[1] = *(const uint4*)(p + 16);
    return f.v;
}

// ---------------------------------------------------------------------------
// Embedding gather + positional add
// ---------------------------------------------------------------------------
__global__ void k_embed(const int* __restrict__ x, const float* __restrict__ emb,
                        const float* __restrict__ pe, float* __restrict__ h) {
    int row = blockIdx.x;
    int tok = x[row];
    for (int d = threadIdx.x; d < DIMM; d += 256)
        h[(size_t)row * DIMM + d] = emb[(size_t)tok * DIMM + d] + pe[(size_t)row * DIMM + d];
}

// g[w,:] = mean over window of token_emb[x] + global_pos_emb[w,:]
__global__ void k_pool(const int* __restrict__ x, const float* __restrict__ emb,
                       const float* __restrict__ gpe, float* __restrict__ g) {
    int w = blockIdx.x;
    for (int d = threadIdx.x; d < DIMM; d += 256) {
        float a = 0.f;
        for (int i = 0; i < WIN; ++i) {
            int tok = x[w * WIN + i];
            a += emb[(size_t)tok * DIMM + d];
        }
        g[(size_t)w * DIMM + d] = a * (1.f / WIN) + gpe[(size_t)w * DIMM + d];
    }
}

// ---------------------------------------------------------------------------
// LayerNorm over last dim (768), emits bf16 (next GEMM's A operand).
// ---------------------------------------------------------------------------
__global__ void k_layernorm(const float* __restrict__ xin, const float* __restrict__ gw,
                            const float* __restrict__ bw, __hip_bfloat16* __restrict__ out) {
    int row = blockIdx.x, tid = threadIdx.x;
    __shared__ float red[256];
    const float* xr = xin + (size_t)row * DIMM;
    float v0 = xr[tid], v1 = xr[tid + 256], v2 = xr[tid + 512];
    red[tid] = v0 + v1 + v2; __syncthreads();
    for (int s = 128; s; s >>= 1) { if (tid < s) red[tid] += red[tid + s]; __syncthreads(); }
    float mu = red[0] * (1.f / DIMM); __syncthreads();
    red[tid] = v0 * v0 + v1 * v1 + v2 * v2; __syncthreads();
    for (int s = 128; s; s >>= 1) { if (tid < s) red[tid] += red[tid + s]; __syncthreads(); }
    float var = red[0] * (1.f / DIMM) - mu * mu;
    float rs = rsqrtf(var + 1e-5f);
    __hip_bfloat16* orow = out + (size_t)row * DIMM;
    orow[tid]       = __float2bfloat16((v0 - mu) * rs * gw[tid]       + bw[tid]);
    orow[tid + 256] = __float2bfloat16((v1 - mu) * rs * gw[tid + 256] + bw[tid + 256]);
    orow[tid + 512] = __float2bfloat16((v2 - mu) * rs * gw[tid + 512] + bw[tid + 512]);
}

// ---------------------------------------------------------------------------
// WMMA GEMM: out[M,N] = epilogue( A[M,K](bf16) @ B[K,N](f32->bf16) )
// Block tile 128x128, BK=32, 256 threads = 8 waves, wave tile 32x64 (2x4 WMMA
// tiles -> 8 v_wmma per k-step per wave). A tile staged with async
// global->LDS b128 copies; B tile converted f32->bf16 through VGPRs.
// Requires: N % 128 == 0, K % 32 == 0. A rows are clamped (not zero-filled):
// out-of-range rows only feed C rows that are never stored.
// ---------------------------------------------------------------------------
__launch_bounds__(256)
__global__ void k_gemm(const __hip_bfloat16* __restrict__ A, const float* __restrict__ B,
                       const float* __restrict__ bias, const float* __restrict__ res,
                       float* __restrict__ outF, __hip_bfloat16* __restrict__ outH,
                       int M, int N, int K, int doGelu) {
    __shared__ __attribute__((aligned(16))) __bf16 sA[128 * 40];  // [m][k], pitch 40
    __shared__ __attribute__((aligned(16))) __bf16 sB[128 * 40];  // [n][k], pitch 40

    int t = threadIdx.x, lane = t & 31, wv = t >> 5;
    int mb = blockIdx.y, nb = blockIdx.x;
    int m0 = (wv >> 1) * 32, n0 = (wv & 1) * 64;

    v8f acc[2][4];
    for (int i = 0; i < 2; ++i)
        for (int j = 0; j < 4; ++j)
            acc[i][j] = (v8f){0.f, 0.f, 0.f, 0.f, 0.f, 0.f, 0.f, 0.f};

    // A staging: thread copies 8 bf16 (16 B at k 0 or 16 of a row of the 128x32 tile), async.
    int ar = t >> 1, akb = (t & 1) * 16;
    int agRow = mb * 128 + ar;
    if (agRow >= M) agRow = M - 1;          // clamp; stores are guarded later
    const __hip_bfloat16* aPtr = A + (size_t)agRow * K + akb;
    __bf16* sap = sA + ar * 40 + akb;
    // B staging: thread converts 16 f32 (column segment of the 32x128 tile).
    int bn = t & 127, bkb = (t >> 7) * 16;
    const float* bPtr = B + (size_t)bkb * N + (size_t)nb * 128 + bn;
    __bf16* sbp = sB + bn * 40 + bkb;

    for (int kt = 0; kt < K; kt += 32) {
        async_g2l_b128(sap, aPtr);                 // global_load_async_to_lds_b128

        union { unsigned short s[8]; uint4 u; } p0, p1;
        for (int j = 0; j < 8; ++j) {
            p0.s[j] = f2bfbits(bPtr[(size_t)j * N]);
            p1.s[j] = f2bfbits(bPtr[(size_t)(j + 8) * N]);
        }
        *(uint4*)(sbp)     = p0.u;
        *(uint4*)(sbp + 8) = p1.u;

        if (kt + 32 < K) {                          // next B tile toward L2
            __builtin_prefetch(bPtr + (size_t)32 * N, 0, 1);
            __builtin_prefetch(aPtr + 32, 0, 1);
        }
        aPtr += 32;
        bPtr += (size_t)32 * N;

        async_wait0();                              // s_wait_asynccnt 0
        __syncthreads();

        v16bf a0 = ldsFrag(sA + (m0)      * 40, 40, lane);
        v16bf a1 = ldsFrag(sA + (m0 + 16) * 40, 40, lane);
        for (int j = 0; j < 4; ++j) {
            v16bf bj = ldsFrag(sB + (n0 + 16 * j) * 40, 40, lane);
            acc[0][j] = bwmma(a0, bj, acc[0][j]);
            acc[1][j] = bwmma(a1, bj, acc[1][j]);
        }
        __syncthreads();
    }

    // Epilogue. C layout: VGPR r -> row r (lanes 0-15) / row 8+r (lanes 16-31).
    int l15 = lane & 15, hf = lane >> 4;
    for (int i = 0; i < 2; ++i) {
        for (int j = 0; j < 4; ++j) {
            int col = nb * 128 + n0 + j * 16 + l15;
            float bi = bias ? bias[col] : 0.f;
            for (int r = 0; r < 8; ++r) {
                int row = mb * 128 + m0 + i * 16 + r + 8 * hf;
                if (row >= M) continue;
                float v = acc[i][j][r] + bi;
                if (doGelu) v = 0.5f * v * (1.f + erff(v * 0.70710678f));
                if (res) v += res[(size_t)row * N + col];
                if (outF) outF[(size_t)row * N + col] = v;
                else      outH[(size_t)row * N + col] = __float2bfloat16(v);
            }
        }
    }
}

// ---------------------------------------------------------------------------
// Local windowed attention. In this configuration the global-token keys are
// FULLY masked by the reference (triu k = ng-W = -97 masks all (bucket, gtok)
// pairs), so they are skipped; bucket 0's "previous window" is zero K/V that
// stays UNmasked (matches the reference's zero-padding semantics).
// One block per (bucket, head); 8 waves; each wave owns 16 query rows.
// ---------------------------------------------------------------------------
__launch_bounds__(256)
__global__ void k_lattn(const __hip_bfloat16* __restrict__ q,
                        const __hip_bfloat16* __restrict__ kv,
                        __hip_bfloat16* __restrict__ ao) {
    extern __shared__ char smem[];
    __bf16* sQ  = (__bf16*)smem;            // [128][72]   query rows
    __bf16* sK  = sQ  + 128 * 72;           // [256][72]   key rows
    __bf16* sVT = sK  + 256 * 72;           // [64][264]   V transposed (d-major)
    __bf16* sP  = sVT + 64 * 264;           // [128][264]  softmax probs

    int w = blockIdx.x, hd = blockIdx.y;
    int t = threadIdx.x, lane = t & 31, wv = t >> 5;
    const int HB = hd * DH;

    // stage Q (128x64 bf16): pure copy -> async global->LDS
    for (int c = t; c < 128 * 8; c += 256) {
        int i = c >> 3, ch = c & 7;
        async_g2l_b128(sQ + i * 72 + ch * 8,
                       q + ((size_t)(w * WIN + i)) * INNER + HB + ch * 8);
    }
    // stage K (256x64): rows 0..127 = previous window (zeros for bucket 0)
    for (int c = t; c < 256 * 8; c += 256) {
        int j = c >> 3, ch = c & 7;
        int pos = w * WIN + j - WIN;
        if (pos >= 0) {
            async_g2l_b128(sK + j * 72 + ch * 8,
                           kv + (size_t)pos * (2 * INNER) + HB + ch * 8);
        } else {
            uint4 z = {0, 0, 0, 0};
            *(uint4*)(sK + j * 72 + ch * 8) = z;
        }
    }
    // stage V transposed: sVT[d][j] (element transpose -> scalar LDS stores)
    for (int c = t; c < 256 * 8; c += 256) {
        int j = c >> 3, ch = c & 7;
        int pos = w * WIN + j - WIN;
        if (pos >= 0) {
            const unsigned short* vp =
                (const unsigned short*)(kv + (size_t)pos * (2 * INNER) + INNER + HB + ch * 8);
            for (int e = 0; e < 8; ++e)
                *(unsigned short*)(sVT + (ch * 8 + e) * 264 + j) = vp[e];
        } else {
            for (int e = 0; e < 8; ++e)
                *(unsigned short*)(sVT + (ch * 8 + e) * 264 + j) = 0;
        }
    }
    async_wait0();            // all async LDS writes landed
    __syncthreads();

    int qbase = wv * 16;
    int l15 = lane & 15, hf = lane >> 4;

    // S = Q @ K^T  (16 rows x 256 cols per wave; 16 accum tiles in registers)
    v8f S[16];
    for (int n = 0; n < 16; ++n)
        S[n] = (v8f){0.f, 0.f, 0.f, 0.f, 0.f, 0.f, 0.f, 0.f};
    for (int kk = 0; kk < 64; kk += 32) {
        v16bf aq = ldsFrag(sQ + qbase * 72 + kk, 72, lane);
        for (int n = 0; n < 16; ++n) {
            v16bf bk = ldsFrag(sK + (n * 16) * 72 + kk, 72, lane);
            S[n] = bwmma(aq, bk, S[n]);
        }
    }

    // scale, causal mask (key col >= query row + W + 1 is future), softmax
    for (int n = 0; n < 16; ++n) {
        int col = n * 16 + l15;
        for (int r = 0; r < 8; ++r) {
            int row = qbase + r + 8 * hf;
            float v = S[n][r] * 0.125f;   // DH^-0.5
            if (col >= row + WIN + 1) v = -3.0e38f;
            S[n][r] = v;
        }
    }
    for (int r = 0; r < 8; ++r) {
        int row = qbase + r + 8 * hf;
        float m = -3.0e38f;
        for (int n = 0; n < 16; ++n) m = fmaxf(m, S[n][r]);
        m = fmaxf(m, __shfl_xor(m, 1)); m = fmaxf(m, __shfl_xor(m, 2));
        m = fmaxf(m, __shfl_xor(m, 4)); m = fmaxf(m, __shfl_xor(m, 8));
        float sum = 0.f;
        for (int n = 0; n < 16; ++n) { float e = __expf(S[n][r] - m); S[n][r] = e; sum += e; }
        sum += __shfl_xor(sum, 1); sum += __shfl_xor(sum, 2);
        sum += __shfl_xor(sum, 4); sum += __shfl_xor(sum, 8);
        float inv = 1.f / sum;
        for (int n = 0; n < 16; ++n)
            sP[row * 264 + n * 16 + l15] = f2bf(S[n][r] * inv);
    }
    // wave reads only its own sP rows; same-wave LDS ops stay in order.

    // O = P @ V   (16 x 64 per wave)
    v8f O[4];
    for (int n = 0; n < 4; ++n)
        O[n] = (v8f){0.f, 0.f, 0.f, 0.f, 0.f, 0.f, 0.f, 0.f};
    for (int kk = 0; kk < 256; kk += 32) {
        v16bf ap = ldsFrag(sP + qbase * 264 + kk, 264, lane);
        for (int n = 0; n < 4; ++n) {
            v16bf bv = ldsFrag(sVT + (n * 16) * 264 + kk, 264, lane);
            O[n] = bwmma(ap, bv, O[n]);
        }
    }
    for (int n = 0; n < 4; ++n) {
        int col = n * 16 + l15;
        for (int r = 0; r < 8; ++r) {
            int row = qbase + r + 8 * hf;
            ao[((size_t)(w * WIN + row)) * INNER + HB + col] = __float2bfloat16(O[n][r]);
        }
    }
}

// ---------------------------------------------------------------------------
// Global (32-token) causal attention — tiny, scalar. One block per head.
// ---------------------------------------------------------------------------
__global__ void k_gattn(const float* __restrict__ qkv, __hip_bfloat16* __restrict__ gao) {
    int hd = blockIdx.x, t = threadIdx.x;
    __shared__ float sQ[32][64], sK[32][64], sV[32][64], sS[32][33];
    for (int e = 0; e < 8; ++e) {
        int idx = t * 8 + e, i = idx >> 6, d = idx & 63;
        sQ[i][d] = qkv[(size_t)i * 1536 + hd * 64 + d] * 0.125f;
        sK[i][d] = qkv[(size_t)i * 1536 + 512 + hd * 64 + d];
        sV[i][d] = qkv[(size_t)i * 1536 + 1024 + hd * 64 + d];
    }
    __syncthreads();
    for (int e = 0; e < 4; ++e) {
        int idx = t * 4 + e, i = idx >> 5, j = idx & 31;
        float s = 0.f;
        for (int d = 0; d < 64; ++d) s += sQ[i][d] * sK[j][d];
        sS[i][j] = (j > i) ? -3.0e38f : s;
    }
    __syncthreads();
    if (t < 32) {
        float m = -3.0e38f;
        for (int j = 0; j < 32; ++j) m = fmaxf(m, sS[t][j]);
        float sum = 0.f;
        for (int j = 0; j < 32; ++j) { float e = __expf(sS[t][j] - m); sS[t][j] = e; sum += e; }
        float inv = 1.f / sum;
        for (int j = 0; j < 32; ++j) sS[t][j] *= inv;
    }
    __syncthreads();
    for (int e = 0; e < 8; ++e) {
        int idx = t * 8 + e, i = idx >> 6, d = idx & 63;
        float o = 0.f;
        for (int j = 0; j < 32; ++j) o += sS[i][j] * sV[j][d];
        gao[(size_t)i * GT_IN + hd * 64 + d] = __float2bfloat16(o);
    }
}

// ---------------------------------------------------------------------------
// Host orchestration
// ---------------------------------------------------------------------------
extern "C" void kernel_launch(void* const* d_in, const int* in_sizes, int n_in,
                              void* d_out, int out_size, void* d_ws, size_t ws_size,
                              hipStream_t stream) {
    const int*   x          = (const int*)  d_in[0];
    const float* token_emb  = (const float*)d_in[1];
    const float* pos_emb    = (const float*)d_in[2];
    const float* gpos_emb   = (const float*)d_in[3];
    const float* gt_ln1_g   = (const float*)d_in[4];
    const float* gt_ln1_b   = (const float*)d_in[5];
    const float* gt_wqkv    = (const float*)d_in[6];
    const float* gt_wo      = (const float*)d_in[7];
    const float* gt_bo      = (const float*)d_in[8];
    const float* gt_ln2_g   = (const float*)d_in[9];
    const float* gt_ln2_b   = (const float*)d_in[10];
    const float* gt_w1      = (const float*)d_in[11];
    const float* gt_b1      = (const float*)d_in[12];
    const float* gt_w2      = (const float*)d_in[13];
    const float* gt_b2      = (const float*)d_in[14];
    const float* ln1_g      = (const float*)d_in[15];
    const float* ln1_b      = (const float*)d_in[16];
    const float* wq         = (const float*)d_in[17];
    const float* wkv        = (const float*)d_in[18];
    const float* wo         = (const float*)d_in[19];
    const float* bo         = (const float*)d_in[20];
    const float* ln2_g      = (const float*)d_in[21];
    const float* ln2_b      = (const float*)d_in[22];
    const float* w1         = (const float*)d_in[23];
    const float* b1         = (const float*)d_in[24];
    const float* w2         = (const float*)d_in[25];
    const float* b2         = (const float*)d_in[26];
    const float* f_ln_g     = (const float*)d_in[27];
    const float* f_ln_b     = (const float*)d_in[28];
    const float* w_logits   = (const float*)d_in[29];
    const float* b_logits   = (const float*)d_in[30];
    (void)in_sizes; (void)n_in; (void)out_size; (void)ws_size;

    char* p = (char*)d_ws;
    auto carve = [&](size_t bytes) { char* r = p; p += (bytes + 255) & ~(size_t)255; return r; };
    float*          h    = (float*)          carve((size_t)SEQ_N * DIMM * 4);
    float*          g    = (float*)          carve((size_t)BKT * DIMM * 4);
    __hip_bfloat16* xln  = (__hip_bfloat16*) carve((size_t)SEQ_N * DIMM * 2);
    __hip_bfloat16* qb   = (__hip_bfloat16*) carve((size_t)SEQ_N * INNER * 2);
    __hip_bfloat16* kvb  = (__hip_bfloat16*) carve((size_t)SEQ_N * 2 * INNER * 2);
    __hip_bfloat16* aob  = (__hip_bfloat16*) carve((size_t)SEQ_N * INNER * 2);
    __hip_bfloat16* t1   = (__hip_bfloat16*) carve((size_t)SEQ_N * FFD * 2);
    __hip_bfloat16* gln  = (__hip_bfloat16*) carve((size_t)BKT * DIMM * 2);
    float*          qkvg = (float*)          carve((size_t)BKT * 3 * GT_IN * 4);
    __hip_bfloat16* gao  = (__hip_bfloat16*) carve((size_t)BKT * GT_IN * 2);
    __hip_bfloat16* gt1  = (__hip_bfloat16*) carve((size_t)BKT * FFD * 2);

    auto gemm = [&](const __hip_bfloat16* A, const float* B, const float* bias,
                    const float* res, float* oF, __hip_bfloat16* oH,
                    int M, int N, int K, int gelu) {
        dim3 grid(N / 128, (M + 127) / 128);
        k_gemm<<<grid, 256, 0, stream>>>(A, B, bias, res, oF, oH, M, N, K, gelu);
    };

    k_embed<<<SEQ_N, 256, 0, stream>>>(x, token_emb, pos_emb, h);
    k_pool<<<BKT, 256, 0, stream>>>(x, token_emb, gpos_emb, g);

    const int lattn_smem = (128 * 72 + 256 * 72 + 64 * 264 + 128 * 264) * 2; // ~153KB

    for (int l = 0; l < DEPTH_L; ++l) {
        // ---- global token path (32 tokens) ----
        k_layernorm<<<BKT, 256, 0, stream>>>(g, gt_ln1_g, gt_ln1_b, gln);
        gemm(gln, gt_wqkv, nullptr, nullptr, qkvg, nullptr, BKT, 3 * GT_IN, DIMM, 0);
        k_gattn<<<GT_H, 256, 0, stream>>>(qkvg, gao);
        gemm(gao, gt_wo, gt_bo, g, g, nullptr, BKT, DIMM, GT_IN, 0);
        k_layernorm<<<BKT, 256, 0, stream>>>(g, gt_ln2_g, gt_ln2_b, gln);
        gemm(gln, gt_w1, gt_b1, nullptr, nullptr, gt1, BKT, FFD, DIMM, 1);
        gemm(gt1, gt_w2, gt_b2, g, g, nullptr, BKT, DIMM, FFD, 0);

        // ---- local token path (4096 tokens) ----
        k_layernorm<<<SEQ_N, 256, 0, stream>>>(h, ln1_g + l * DIMM, ln1_b + l * DIMM, xln);
        gemm(xln, wq + (size_t)l * DIMM * INNER, nullptr, nullptr, nullptr, qb,
             SEQ_N, INNER, DIMM, 0);
        gemm(xln, wkv + (size_t)l * DIMM * 2 * INNER, nullptr, nullptr, nullptr, kvb,
             SEQ_N, 2 * INNER, DIMM, 0);
        k_lattn<<<dim3(BKT, HEADS), 256, lattn_smem, stream>>>(qb, kvb, aob);
        gemm(aob, wo + (size_t)l * INNER * DIMM, bo + l * DIMM, h, h, nullptr,
             SEQ_N, DIMM, INNER, 0);
        k_layernorm<<<SEQ_N, 256, 0, stream>>>(h, ln2_g + l * DIMM, ln2_b + l * DIMM, xln);
        gemm(xln, w1 + (size_t)l * DIMM * FFD, b1 + l * FFD, nullptr, nullptr, t1,
             SEQ_N, FFD, DIMM, 1);
        gemm(t1, w2 + (size_t)l * FFD * DIMM, b2 + l * DIMM, h, h, nullptr,
             SEQ_N, DIMM, FFD, 0);
    }

    // ---- final LN + LM head (201 GFLOP; w_logits f32 stays L2-resident) ----
    k_layernorm<<<SEQ_N, 256, 0, stream>>>(h, f_ln_g, f_ln_b, xln);
    gemm(xln, w_logits, b_logits, nullptr, (float*)d_out, nullptr,
         SEQ_N, 32000, DIMM, 0);
}
